// MultiHeadHierarchicalAttention_6262062318244
// MI455X (gfx1250) — matrix operand order repaired and checked by reading, not compiled, not run on hardware
//
#include <hip/hip_runtime.h>
#include <hip/hip_bf16.h>

// ---------------------------------------------------------------------------
// Types for CDNA5 WMMA (wave32): v_wmma_f32_16x16x32_bf16
// ---------------------------------------------------------------------------
typedef __bf16 bf16_t;
typedef bf16_t v16bf __attribute__((ext_vector_type(16)));
typedef bf16_t v8bf  __attribute__((ext_vector_type(8)));
typedef float  v8f   __attribute__((ext_vector_type(8)));

// Dims
#define Bv_   4
#define LQ_   256
#define NB_   32
#define NT_   128
#define NTOP_ 100
#define D_    512
#define H_    8
#define DK_   64
#define SCALE_ 0.125f   // DK^-0.5

__device__ __forceinline__ bf16_t to_bf(float x) { return (bf16_t)x; }

// A-fragment (16x32 bf16) from row-major LDS tile [rows][stride] at (m,k).
// ISA 7.12.2: lanes 0-15 row M=lane, halves K=8g..8g+7 and 16+8g..16+8g+7.
__device__ __forceinline__ v16bf ld_afrag(const bf16_t* tile, int stride, int m, int k) {
  int lane = threadIdx.x & 31;
  int r = lane & 15, g = lane >> 4;
  const bf16_t* p = tile + (size_t)(m + r) * stride + k + 8 * g;
  v8bf lo = *(const v8bf*)(p);
  v8bf hi = *(const v8bf*)(p + 16);
  v16bf a;
#pragma unroll
  for (int i = 0; i < 8; i++) { a[i] = lo[i]; a[i + 8] = hi[i]; }
  return a;
}

// B-fragment (32x16 bf16) from LDS stored as Wt[n][k] row-major [N][stride].
// Lane: n = lane%16, halves K=8g..8g+7 and 16+8g..16+8g+7 (matches A).
__device__ __forceinline__ v16bf ld_bfrag(const bf16_t* tileT, int stride, int n, int k) {
  int lane = threadIdx.x & 31;
  int c = lane & 15, g = lane >> 4;
  const bf16_t* p = tileT + (size_t)(n + c) * stride + k + 8 * g;
  v8bf lo = *(const v8bf*)(p);
  v8bf hi = *(const v8bf*)(p + 16);
  v16bf b;
#pragma unroll
  for (int i = 0; i < 8; i++) { b[i] = lo[i]; b[i + 8] = hi[i]; }
  return b;
}

__device__ __forceinline__ v8f wmma_bf16(v16bf a, v16bf b, v8f c) {
  return __builtin_amdgcn_wmma_f32_16x16x32_bf16(false, a, false, b, (short)0, c,
                                                 false, false);
}

// ---- vectorized staging helpers (B128 in, B128/B16 out) -------------------
// rows x 64 f32 (row stride srs) -> LDS bf16 [rows][dstride]
__device__ __forceinline__ void stage64(bf16_t* dst, int dstride,
                                        const float* src, int srs, int rows) {
  for (int idx = threadIdx.x; idx < rows * 8; idx += 256) {
    int r = idx >> 3, c8 = (idx & 7) * 8;
    const float* p = src + (size_t)r * srs + c8;
    float4 x = *(const float4*)p;
    float4 y = *(const float4*)(p + 4);
    v8bf pk;
    pk[0] = to_bf(x.x); pk[1] = to_bf(x.y); pk[2] = to_bf(x.z); pk[3] = to_bf(x.w);
    pk[4] = to_bf(y.x); pk[5] = to_bf(y.y); pk[6] = to_bf(y.z); pk[7] = to_bf(y.w);
    *(v8bf*)(dst + (size_t)r * dstride + c8) = pk;
  }
}
// rows x 64 f32 -> transposed LDS bf16 dst[c][r], c in [0,64)
__device__ __forceinline__ void stage64T(bf16_t* dst, int dstride,
                                         const float* src, int srs, int rows) {
  for (int idx = threadIdx.x; idx < rows * 16; idx += 256) {
    int r = idx >> 4, c4 = (idx & 15) * 4;
    float4 x = *(const float4*)(src + (size_t)r * srs + c4);
    dst[(size_t)(c4 + 0) * dstride + r] = to_bf(x.x);
    dst[(size_t)(c4 + 1) * dstride + r] = to_bf(x.y);
    dst[(size_t)(c4 + 2) * dstride + r] = to_bf(x.z);
    dst[(size_t)(c4 + 3) * dstride + r] = to_bf(x.w);
  }
}
__device__ __forceinline__ void zero_lds(bf16_t* dst, int halfs) {  // halfs % 8 == 0
  v8bf z = {};
  for (int idx = threadIdx.x; idx < (halfs >> 3); idx += 256)
    *(v8bf*)(dst + idx * 8) = z;
}

// ---------------------------------------------------------------------------
// Weight pre-transpose: W [K,N] f32 -> WT [N,K] bf16 (done once per weight)
// ---------------------------------------------------------------------------
__global__ __launch_bounds__(256) void wtrans_kernel(const float* __restrict__ W,
                                                     bf16_t* __restrict__ WT,
                                                     int K, int N) {
  __shared__ float tile[32][33];
  int k0 = blockIdx.x * 32, n0 = blockIdx.y * 32;
  int t = threadIdx.x;
#pragma unroll
  for (int j = 0; j < 4; j++) {
    int e = t + j * 256;
    tile[e >> 5][e & 31] = W[(size_t)(k0 + (e >> 5)) * N + n0 + (e & 31)];
  }
  __syncthreads();
  if (t < 128) {
    int n = t >> 2, k8 = (t & 3) * 8;
    v8bf pk;
#pragma unroll
    for (int i = 0; i < 8; i++) pk[i] = to_bf(tile[k8 + i][n]);
    *(v8bf*)(WT + (size_t)(n0 + n) * K + k0 + k8) = pk;
  }
}

// ---------------------------------------------------------------------------
// GEMM + bias:  C[m, coff+n] = sum_k A[m,k] * W[k,n] + bias[n]
// A: [M, lda] f32; WT: [N, K] bf16 (pre-transposed). 64x64 tile, 8 waves.
// ---------------------------------------------------------------------------
__global__ __launch_bounds__(256) void gemm_bias_kernel(
    const float* __restrict__ A, int lda,
    const bf16_t* __restrict__ WT,
    const float* __restrict__ bias,
    float* __restrict__ C, int ldc, int coff,
    int M, int N, int K) {
  __shared__ __align__(16) bf16_t As[64 * 40];
  __shared__ __align__(16) bf16_t Ws[64 * 40];

  int m0 = blockIdx.x * 64, n0 = blockIdx.y * 64;
  int t = threadIdx.x;
  int w = t >> 5;
  int mW = (w & 3) * 16, nW = (w >> 2) * 32;

  int arow = t >> 2, ac8 = (t & 3) * 8;         // 4 threads per tile row
  bool aok = (m0 + arow) < M;
  const float*  asrc = A  + (size_t)(aok ? (m0 + arow) : (M - 1)) * lda + ac8;
  const bf16_t* wsrc = WT + (size_t)(n0 + arow) * K + ac8;
  bf16_t* adst = As + arow * 40 + ac8;
  bf16_t* wdst = Ws + arow * 40 + ac8;

  v8f acc0 = {}; v8f acc1 = {};

  for (int kc = 0; kc < K; kc += 32) {
    __syncthreads();
    float4 a0 = *(const float4*)(asrc + kc);
    float4 a1 = *(const float4*)(asrc + kc + 4);
    v8bf wv = *(const v8bf*)(wsrc + kc);
    if (kc + 32 < K) __builtin_prefetch(asrc + kc + 32, 0, 1);
    v8bf pk;
    pk[0] = to_bf(aok ? a0.x : 0.f); pk[1] = to_bf(aok ? a0.y : 0.f);
    pk[2] = to_bf(aok ? a0.z : 0.f); pk[3] = to_bf(aok ? a0.w : 0.f);
    pk[4] = to_bf(aok ? a1.x : 0.f); pk[5] = to_bf(aok ? a1.y : 0.f);
    pk[6] = to_bf(aok ? a1.z : 0.f); pk[7] = to_bf(aok ? a1.w : 0.f);
    *(v8bf*)adst = pk;
    *(v8bf*)wdst = wv;
    __syncthreads();

    v16bf a = ld_afrag(As, 40, mW, 0);
    acc0 = wmma_bf16(a, ld_bfrag(Ws, 40, nW, 0), acc0);
    acc1 = wmma_bf16(a, ld_bfrag(Ws, 40, nW + 16, 0), acc1);
  }

  int lane = t & 31, cn = lane & 15, g = lane >> 4;
  float bv0 = bias ? bias[n0 + nW + cn] : 0.f;
  float bv1 = bias ? bias[n0 + nW + 16 + cn] : 0.f;
#pragma unroll
  for (int i = 0; i < 8; i++) {
    int m = m0 + mW + g * 8 + i;
    if (m < M) {
      C[(size_t)m * ldc + coff + n0 + nW + cn]      = acc0[i] + bv0;
      C[(size_t)m * ldc + coff + n0 + nW + 16 + cn] = acc1[i] + bv1;
    }
  }
}

// ---------------------------------------------------------------------------
// pos = NB * sigmoid(fc2(tanh(fc1(qs_row))))   one 64-thread block per row
// ---------------------------------------------------------------------------
__global__ __launch_bounds__(64) void pos_kernel(
    const float* __restrict__ qs, const float* __restrict__ w1,
    const float* __restrict__ b1, const float* __restrict__ w2,
    const float* __restrict__ b2, float* __restrict__ pos) {
  int idx = blockIdx.x;                 // b*H*LQ + h*LQ + q
  int q = idx & (LQ_ - 1);
  int h = (idx >> 8) & (H_ - 1);
  int b = idx >> 11;
  const float* row = qs + ((size_t)(b * LQ_ + q)) * D_ + h * DK_;
  __shared__ float r[64];
  __shared__ float t1[64];
  int j = threadIdx.x;
  r[j] = row[j];
  __syncthreads();
  float s = b1[j];
#pragma unroll 8
  for (int k = 0; k < 64; k++) s += r[k] * w1[k * 64 + j];
  t1[j] = tanhf(s) * w2[j];
  for (int o = 32; o > 0; o >>= 1) { __syncthreads(); if (j < o) t1[j] += t1[j + o]; }
  __syncthreads();
  if (j == 0) {
    float z = t1[0] + b2[0];
    pos[idx] = (float)NB_ * (1.f / (1.f + __expf(-z)));
  }
}

// ---------------------------------------------------------------------------
// Sentence attention. One block per (b,h,qchunk of 128).
// ---------------------------------------------------------------------------
__global__ __launch_bounds__(256) void sent_attn_kernel(
    const float* __restrict__ qs, const float* __restrict__ ks,
    const float* __restrict__ ptp, const float* __restrict__ vs,
    const float* __restrict__ bias_s, const float* __restrict__ pos,
    float* __restrict__ attn_s, float* __restrict__ ctx) {
  int bid = blockIdx.x;
  int qc = bid & 1, h = (bid >> 1) & 7, b = bid >> 4;
  int q0 = qc * 128;
  __shared__ __align__(16) bf16_t Aq[128 * 72];
  __shared__ __align__(16) bf16_t KPT[32 * 72];
  __shared__ __align__(16) bf16_t VsT[64 * 40];
  __shared__ __align__(16) bf16_t Pa[128 * 40];
  int t = threadIdx.x;

  stage64(Aq, 72, qs + ((size_t)(b * LQ_ + q0)) * D_ + h * DK_, D_, 128);
  for (int idx = t; idx < 32 * 8; idx += 256) {          // (ks + pt) 32x64
    int r = idx >> 3, c8 = (idx & 7) * 8;
    size_t off = ((size_t)(b * NB_ + r)) * D_ + h * DK_ + c8;
    float4 x = *(const float4*)(ks + off);
    float4 y = *(const float4*)(ks + off + 4);
    float4 u = *(const float4*)(ptp + off);
    float4 v = *(const float4*)(ptp + off + 4);
    v8bf pk;
    pk[0] = to_bf(x.x + u.x); pk[1] = to_bf(x.y + u.y);
    pk[2] = to_bf(x.z + u.z); pk[3] = to_bf(x.w + u.w);
    pk[4] = to_bf(y.x + v.x); pk[5] = to_bf(y.y + v.y);
    pk[6] = to_bf(y.z + v.z); pk[7] = to_bf(y.w + v.w);
    *(v8bf*)(KPT + r * 72 + c8) = pk;
  }
  stage64T(VsT, 40, vs + ((size_t)(b * NB_)) * D_ + h * DK_, D_, 32);
  __syncthreads();

  int w = t >> 5, lane = t & 31, cn = lane & 15, g = lane >> 4;
  v8f s0 = {}, s1 = {};
#pragma unroll
  for (int kc = 0; kc < 64; kc += 32) {
    v16bf a = ld_afrag(Aq, 72, 16 * w, kc);
    s0 = wmma_bf16(a, ld_bfrag(KPT, 72, 0, kc), s0);
    s1 = wmma_bf16(a, ld_bfrag(KPT, 72, 16, kc), s1);
  }
  size_t base = (((size_t)b * H_ + h) * LQ_ + q0);
#pragma unroll
  for (int i = 0; i < 8; i++) {
    int m = 16 * w + 8 * g + i;
    float pc = pos[base + m];
    float c0 = (float)cn, c1 = (float)(cn + 16);
    float v0 = s0[i] * SCALE_ + bias_s[(base + m) * NB_ + cn]      - (c0 - pc) * (c0 - pc) * 0.5f;
    float v1 = s1[i] * SCALE_ + bias_s[(base + m) * NB_ + cn + 16] - (c1 - pc) * (c1 - pc) * 0.5f;
    float mx = fmaxf(v0, v1);
    for (int o = 1; o < 16; o <<= 1) mx = fmaxf(mx, __shfl_xor(mx, o, 16));
    float e0 = __expf(v0 - mx), e1 = __expf(v1 - mx);
    float sm = e0 + e1;
    for (int o = 1; o < 16; o <<= 1) sm += __shfl_xor(sm, o, 16);
    float inv = 1.f / sm;
    e0 *= inv; e1 *= inv;
    attn_s[(base + m) * NB_ + cn]      = e0;
    attn_s[(base + m) * NB_ + cn + 16] = e1;
    Pa[m * 40 + cn]      = to_bf(e0);
    Pa[m * 40 + cn + 16] = to_bf(e1);
  }
  __syncthreads();

  v8f c[4] = {};
  {
    v16bf a = ld_afrag(Pa, 40, 16 * w, 0);
#pragma unroll
    for (int f = 0; f < 4; f++) c[f] = wmma_bf16(a, ld_bfrag(VsT, 40, 16 * f, 0), c[f]);
  }
#pragma unroll
  for (int f = 0; f < 4; f++)
#pragma unroll
    for (int i = 0; i < 8; i++) {
      int m = 16 * w + 8 * g + i;
      ctx[((size_t)(b * LQ_ + q0 + m)) * D_ + h * DK_ + 16 * f + cn] = c[f][i];
    }
}

// ---------------------------------------------------------------------------
// Topic attention. NTOP=100 padded to 112 score cols / 128 K for ctx.
// Q fragments kept in registers; LDS overlays: Y = qt -> attn, X = kt^T -> vt^T.
// ---------------------------------------------------------------------------
__global__ __launch_bounds__(256) void topic_attn_kernel(
    const float* __restrict__ qt, const float* __restrict__ kt,
    const float* __restrict__ vt, const float* __restrict__ tbias,
    float* __restrict__ ctx) {
  int bid = blockIdx.x;
  int qc = bid & 1, h = (bid >> 1) & 7, b = bid >> 4;
  int q0 = qc * 128;
  __shared__ __align__(16) bf16_t Y[128 * 136];   // qt[128][72] -> attn[128][136]
  __shared__ __align__(16) bf16_t X[64 * 136];    // ktT[112][72] -> vtT[64][136]
  int t = threadIdx.x;
  int w = t >> 5, lane = t & 31, cn = lane & 15, g = lane >> 4;

  stage64(Y, 72, qt + ((size_t)(b * LQ_ + q0)) * D_ + h * DK_, D_, 128);
  zero_lds(X, 112 * 72);
  __syncthreads();
  stage64(X, 72, kt + ((size_t)(b * NTOP_)) * D_ + h * DK_, D_, NTOP_);
  __syncthreads();

  v16bf aq0 = ld_afrag(Y, 72, 16 * w, 0);
  v16bf aq1 = ld_afrag(Y, 72, 16 * w, 32);
  v8f s[7] = {};
#pragma unroll
  for (int tt = 0; tt < 7; tt++) {
    s[tt] = wmma_bf16(aq0, ld_bfrag(X, 72, 16 * tt, 0), s[tt]);
    s[tt] = wmma_bf16(aq1, ld_bfrag(X, 72, 16 * tt, 32), s[tt]);
  }
  size_t base = (((size_t)b * H_ + h) * LQ_ + q0);
#pragma unroll
  for (int i = 0; i < 8; i++) {
    int m = 16 * w + 8 * g + i;
    float vv[7]; float mx = -3.0e38f;
#pragma unroll
    for (int tt = 0; tt < 7; tt++) {
      int col = 16 * tt + cn;
      float v = (col < NTOP_)
                    ? s[tt][i] * SCALE_ + tbias[(base + m) * NTOP_ + col]
                    : -1.0e30f;
      vv[tt] = v; mx = fmaxf(mx, v);
    }
    for (int o = 1; o < 16; o <<= 1) mx = fmaxf(mx, __shfl_xor(mx, o, 16));
    float sm = 0.f;
#pragma unroll
    for (int tt = 0; tt < 7; tt++) { vv[tt] = __expf(vv[tt] - mx); sm += vv[tt]; }
    for (int o = 1; o < 16; o <<= 1) sm += __shfl_xor(sm, o, 16);
    float inv = 1.f / sm;
#pragma unroll
    for (int tt = 0; tt < 7; tt++) s[tt][i] = vv[tt] * inv;   // stash in acc regs
  }
  __syncthreads();
  zero_lds(Y, 128 * 136);                 // pads cols >= 112
  zero_lds(X, 64 * 136);                  // pads K >= 100
  __syncthreads();
  stage64T(X, 136, vt + ((size_t)(b * NTOP_)) * D_ + h * DK_, D_, NTOP_);
#pragma unroll
  for (int i = 0; i < 8; i++) {
    int m = 16 * w + 8 * g + i;
#pragma unroll
    for (int tt = 0; tt < 7; tt++) Y[m * 136 + 16 * tt + cn] = to_bf(s[tt][i]);
  }
  __syncthreads();

  v8f c[4] = {};
#pragma unroll
  for (int kc = 0; kc < 128; kc += 32) {
    v16bf a = ld_afrag(Y, 136, 16 * w, kc);
#pragma unroll
    for (int f = 0; f < 4; f++) c[f] = wmma_bf16(a, ld_bfrag(X, 136, 16 * f, kc), c[f]);
  }
#pragma unroll
  for (int f = 0; f < 4; f++)
#pragma unroll
    for (int i = 0; i < 8; i++) {
      int m = 16 * w + 8 * g + i;
      ctx[((size_t)(b * LQ_ + q0 + m)) * D_ + h * DK_ + 16 * f + cn] = c[f][i];
    }
}

// ---------------------------------------------------------------------------
// Word-level hierarchical attention. One block per (b,h,qchunk of 128),
// NB-loop with persistent f32 WMMA accumulators; qw fragments in registers.
// ---------------------------------------------------------------------------
__global__ __launch_bounds__(256) void word_attn_kernel(
    const float* __restrict__ qw, const float* __restrict__ kw,
    const float* __restrict__ vw, const float* __restrict__ bias_w,
    const float* __restrict__ attn_s, float* __restrict__ ctx) {
  int bid = blockIdx.x;
  int qc = bid & 1, h = (bid >> 1) & 7, b = bid >> 4;
  int q0 = qc * 128;
  __shared__ __align__(16) bf16_t Y[128 * 136];   // qw once -> attn per block
  __shared__ __align__(16) bf16_t X[128 * 72];    // kw -> vw^T per block
  int t = threadIdx.x;
  int w = t >> 5, lane = t & 31, cn = lane & 15, g = lane >> 4;
  size_t base = (((size_t)b * H_ + h) * LQ_ + q0);

  stage64(Y, 72, qw + ((size_t)(b * LQ_ + q0)) * D_ + h * DK_, D_, 128);
  __syncthreads();
  v16bf aq0 = ld_afrag(Y, 72, 16 * w, 0);   // Q fragments live in registers
  v16bf aq1 = ld_afrag(Y, 72, 16 * w, 32);

  v8f c[4] = {};   // ctx accumulators across paragraph blocks

  for (int nb = 0; nb < NB_; nb++) {
    __syncthreads();                             // prev ctx wmma done with X/Y
    stage64(X, 72, kw + ((size_t)((b * NB_ + nb) * NT_)) * D_ + h * DK_, D_, 128);
    __syncthreads();

    v8f s[8] = {};
#pragma unroll
    for (int tt = 0; tt < 8; tt++) {
      s[tt] = wmma_bf16(aq0, ld_bfrag(X, 72, 16 * tt, 0), s[tt]);
      s[tt] = wmma_bf16(aq1, ld_bfrag(X, 72, 16 * tt, 32), s[tt]);
    }

    size_t bwbase = (((size_t)(b * NB_ + nb) * H_ + h) * LQ_ + q0) * NT_;
#pragma unroll
    for (int i = 0; i < 8; i++) {
      int m = 16 * w + 8 * g + i;
      float vv[8]; float mx = -3.0e38f;
#pragma unroll
      for (int tt = 0; tt < 8; tt++) {
        float v = s[tt][i] * SCALE_ + bias_w[bwbase + (size_t)m * NT_ + 16 * tt + cn];
        vv[tt] = v; mx = fmaxf(mx, v);
      }
      for (int o = 1; o < 16; o <<= 1) mx = fmaxf(mx, __shfl_xor(mx, o, 16));
      float sm = 0.f;
#pragma unroll
      for (int tt = 0; tt < 8; tt++) { vv[tt] = __expf(vv[tt] - mx); sm += vv[tt]; }
      for (int o = 1; o < 16; o <<= 1) sm += __shfl_xor(sm, o, 16);
      float aw = attn_s[(base + m) * NB_ + nb];   // hierarchical reweight
      float inv = aw / sm;
#pragma unroll
      for (int tt = 0; tt < 8; tt++) s[tt][i] = vv[tt] * inv;
    }
    __syncthreads();                             // scores done reading X(kw)
#pragma unroll
    for (int i = 0; i < 8; i++) {                // attn_hier -> Y [128][136]
      int m = 16 * w + 8 * g + i;
#pragma unroll
      for (int tt = 0; tt < 8; tt++) Y[m * 136 + 16 * tt + cn] = to_bf(s[tt][i]);
    }
    stage64T(X, 136, vw + ((size_t)((b * NB_ + nb) * NT_)) * D_ + h * DK_, D_, 128);
    __syncthreads();
#pragma unroll
    for (int kc = 0; kc < 128; kc += 32) {
      v16bf a = ld_afrag(Y, 136, 16 * w, kc);
#pragma unroll
      for (int f = 0; f < 4; f++) c[f] = wmma_bf16(a, ld_bfrag(X, 136, 16 * f, kc), c[f]);
    }
  }

#pragma unroll
  for (int f = 0; f < 4; f++)
#pragma unroll
    for (int i = 0; i < 8; i++) {
      int m = 16 * w + 8 * g + i;
      ctx[((size_t)(b * LQ_ + q0 + m)) * D_ + h * DK_ + 16 * f + cn] = c[f][i];
    }
}

// ---------------------------------------------------------------------------
// Host launcher
// ---------------------------------------------------------------------------
extern "C" void kernel_launch(void* const* d_in, const int* in_sizes, int n_in,
                              void* d_out, int out_size, void* d_ws, size_t ws_size,
                              hipStream_t stream) {
  const float* q_in    = (const float*)d_in[0];
  const float* k_s_in  = (const float*)d_in[1];
  const float* v_s_in  = (const float*)d_in[2];
  const float* k_w_in  = (const float*)d_in[3];
  const float* v_w_in  = (const float*)d_in[4];
  const float* bias_w  = (const float*)d_in[5];
  const float* bias_s  = (const float*)d_in[6];
  // d_in[7] = graph_attn_bias (unused by the forward pass)
  const float* topic   = (const float*)d_in[8];
  const float* tbias   = (const float*)d_in[9];
  const float* pt_in   = (const float*)d_in[10];
  auto Wp = [&](int i) { return (const float*)d_in[11 + 2 * i]; };
  auto Bp = [&](int i) { return (const float*)d_in[12 + 2 * i]; };
  // 0 w_qs_s 1 w_ks_s 2 w_vs_s 3 w_qs_w 4 w_ks_w 5 w_vs_w 6 w_qs_t 7 w_ks_t
  // 8 w_vs_t 9 fc_pt 10 fc_topic 11 proj_s 12 proj_w 13 fc 14 pos_fc1 15 pos_fc2

  float* p = (float*)d_ws;
  auto take = [&](size_t n) { float* r = p; p += n; return r; };
  const size_t MQ = (size_t)Bv_ * LQ_;          // 1024
  const size_t MS = (size_t)Bv_ * NB_;          // 128
  const size_t MW = (size_t)Bv_ * NB_ * NT_;    // 16384
  const size_t MT = (size_t)Bv_ * NTOP_;        // 400
  float* qs   = take(MQ * D_);
  float* qw   = take(MQ * D_);
  float* qt   = take(MQ * D_);
  float* ksb  = take(MS * D_);
  float* vsb  = take(MS * D_);
  float* ptb  = take(MS * D_);
  float* kwb  = take(MW * D_);
  float* vwb  = take(MW * D_);
  float* ktb  = take(MT * D_);
  float* vtb  = take(MT * D_);
  float* pos  = take((size_t)Bv_ * H_ * LQ_);
  float* atns = take((size_t)Bv_ * H_ * LQ_ * NB_);
  float* ctxs = take(MQ * D_);
  float* ctxw = take(MQ * D_);
  float* ctxt = take(MQ * D_);
  float* cat  = take(MQ * 3 * D_);
  // bf16 pre-transposed weights: 13 x [512][512] + fc [512][1536]
  bf16_t* wtb = (bf16_t*)p;
  bf16_t* WT[14];
  for (int i = 0; i < 14; i++) WT[i] = wtb + (size_t)i * D_ * D_;  // fc extends past
  (void)ws_size; (void)in_sizes; (void)n_in; (void)out_size;

  dim3 blk(256);
  // Pre-transpose + convert all weights once
  for (int i = 0; i < 13; i++)
    wtrans_kernel<<<dim3(D_ / 32, D_ / 32), blk, 0, stream>>>(Wp(i), WT[i], D_, D_);
  wtrans_kernel<<<dim3(3 * D_ / 32, D_ / 32), blk, 0, stream>>>(Wp(13), WT[13], 3 * D_, D_);

  auto gemm = [&](const float* A, int lda, const bf16_t* Wm, const float* bv,
                  float* C, int ldc, int coff, int M, int N, int K) {
    dim3 grid((M + 63) / 64, N / 64);
    gemm_bias_kernel<<<grid, blk, 0, stream>>>(A, lda, Wm, bv, C, ldc, coff, M, N, K);
  };

  // Projections
  gemm(q_in,   D_, WT[0],  Bp(0),  qs,  D_, 0, (int)MQ, D_, D_);   // w_qs_s
  gemm(k_s_in, D_, WT[1],  Bp(1),  ksb, D_, 0, (int)MS, D_, D_);   // w_ks_s
  gemm(v_s_in, D_, WT[2],  Bp(2),  vsb, D_, 0, (int)MS, D_, D_);   // w_vs_s
  gemm(pt_in,  D_, WT[9],  Bp(9),  ptb, D_, 0, (int)MS, D_, D_);   // fc_pt
  gemm(q_in,   D_, WT[3],  Bp(3),  qw,  D_, 0, (int)MQ, D_, D_);   // w_qs_w
  gemm(k_w_in, D_, WT[4],  Bp(4),  kwb, D_, 0, (int)MW, D_, D_);   // w_ks_w
  gemm(v_w_in, D_, WT[5],  Bp(5),  vwb, D_, 0, (int)MW, D_, D_);   // w_vs_w
  gemm(q_in,   D_, WT[6],  Bp(6),  qt,  D_, 0, (int)MQ, D_, D_);   // w_qs_t
  gemm(topic,  D_, WT[7],  Bp(7),  ktb, D_, 0, (int)MT, D_, D_);   // w_ks_t
  gemm(topic,  D_, WT[8],  Bp(8),  vtb, D_, 0, (int)MT, D_, D_);   // w_vs_t

  // Position prior
  pos_kernel<<<dim3(Bv_ * H_ * LQ_), dim3(64), 0, stream>>>(
      qs, Wp(14), Bp(14), Wp(15), Bp(15), pos);

  // Attention stages
  dim3 agrid(Bv_ * H_ * 2);
  sent_attn_kernel<<<agrid, blk, 0, stream>>>(qs, ksb, ptb, vsb, bias_s, pos,
                                              atns, ctxs);
  word_attn_kernel<<<agrid, blk, 0, stream>>>(qw, kwb, vwb, bias_w, atns, ctxw);
  topic_attn_kernel<<<agrid, blk, 0, stream>>>(qt, ktb, vtb, tbias, ctxt);

  // Output projections into concat [ctx_t | ctx_s | ctx_w], then fc
  gemm(ctxt, D_, WT[10], Bp(10), cat, 3 * D_, 0,      (int)MQ, D_, D_); // fc_topic
  gemm(ctxs, D_, WT[11], Bp(11), cat, 3 * D_, D_,     (int)MQ, D_, D_); // proj_s
  gemm(ctxw, D_, WT[12], Bp(12), cat, 3 * D_, 2 * D_, (int)MQ, D_, D_); // proj_w
  gemm(cat, 3 * D_, WT[13], Bp(13), (float*)d_out, D_, 0, (int)MQ, D_, 3 * D_); // fc
}
